// SEGNO_GCL_75591424410042
// MI455X (gfx1250) — matrix operand order repaired and checked by reading, not compiled, not run on hardware
//
#include <hip/hip_runtime.h>

typedef __attribute__((ext_vector_type(16))) _Float16 v16h;
typedef __attribute__((ext_vector_type(8)))  _Float16 v8h;
typedef __attribute__((ext_vector_type(8)))  float    v8f;

#define F 64
#define EPB 128
#define A1_STRIDE 168   // halfs; 336B row stride, 16B aligned, bank-skewed
#define HB_STRIDE 72    // halfs; 144B row stride, 16B aligned, bank-skewed
#define AN_STRIDE 136   // halfs; 272B row stride, 16B aligned, bank-skewed

// ---------- WMMA fragment helpers (CDNA5 16x16x32 f16 layouts) ----------

__device__ __forceinline__ v16h cmb(v8h lo, v8h hi) {
  v16h r;
#pragma unroll
  for (int i = 0; i < 8; i++) { r[i] = lo[i]; r[i + 8] = hi[i]; }
  return r;
}

// A (16x32, f16): lane row M = lane&15; g=lane>>4 selects K-half.
// halves 0..7 -> K = 8g + [0..7]; halves 8..15 -> K = 8g + 16 + [0..7]
__device__ __forceinline__ v16h load_a(const _Float16* A, int stride, int m, int g, int kbase) {
  const _Float16* p = A + m * stride + kbase + g * 8;
  return cmb(*(const v8h*)p, *(const v8h*)(p + 16));
}

// B (32x16, f16) from transposed weights wt[n][k]: lane col N = lane&15,
// halves 0..7 -> K = 8g + [0..7]; halves 8..15 -> K = 8g + 16 + [0..7]
__device__ __forceinline__ v16h load_b(const _Float16* wt, int Kd, int n, int g, int kbase) {
  const _Float16* p = wt + (size_t)n * Kd + kbase + g * 8;
  return cmb(*(const v8h*)p, *(const v8h*)(p + 16));
}

__device__ __forceinline__ v8f wmma_f16(v16h a, v16h b, v8f c) {
  return __builtin_amdgcn_wmma_f32_16x16x32_f16(false, a, false, b, (short)0, c, false, false);
}

// 16xK A-tile (LDS) times Kx64 weights (global, transposed) -> 4 accumulators
template <int S>
__device__ __forceinline__ void gemm_tile(const _Float16* Abase, int astride,
                                          const _Float16* wt, int Kd,
                                          int lane, v8f acc[4]) {
  int m = lane & 15, g = lane >> 4;
#pragma unroll
  for (int s = 0; s < S; s++) {
    v16h a = load_a(Abase, astride, m, g, s * 32);
#pragma unroll
    for (int t = 0; t < 4; t++) {
      v16h b = load_b(wt, Kd, (lane & 15) + 16 * t, g, s * 32);
      acc[t] = wmma_f16(a, b, acc[t]);
    }
  }
}

// bias + ReLU on f32 accumulators, store as f16 A-tile for the next layer.
// C/D layout: lane -> N = lane&15 (+16 per tile), VGPR i -> M = i + 8*(lane>>4)
__device__ __forceinline__ void relu_store(v8f acc[4], const float* bias,
                                           _Float16* Hbase, int stride, int lane) {
  int g = lane >> 4, nl = lane & 15;
#pragma unroll
  for (int t = 0; t < 4; t++) {
    float bb = bias[nl + 16 * t];
#pragma unroll
    for (int i = 0; i < 8; i++) {
      float v = acc[t][i] + bb;
      v = v > 0.f ? v : 0.f;
      Hbase[(i + 8 * g) * stride + nl + 16 * t] = (_Float16)v;
    }
  }
}

// ---------- workspace prep ----------

__global__ void segno_zero_ws(float* __restrict__ p, int n) {
  int i = blockIdx.x * blockDim.x + threadIdx.x;
  int stride = gridDim.x * blockDim.x;
  for (; i < n; i += stride) p[i] = 0.f;
}

// transposed f16 weights, tightly packed in one half-precision arena:
// wt1 [64][160] @0 (K padded 129->160 with zeros), wt2 [64][64] @10240,
// wc1t [64][64] @14336, wn1t [64][128] @18432, wn2t [64][64] @26624
__global__ void segno_prep_w(const float* __restrict__ We1, const float* __restrict__ We2,
                             const float* __restrict__ Wc1, const float* __restrict__ Wn1,
                             const float* __restrict__ Wn2, _Float16* __restrict__ w) {
  int i = blockIdx.x * blockDim.x + threadIdx.x;
  if (i < 10240)      { int n = i / 160, k = i % 160; w[i] = (k < 129) ? (_Float16)We1[k * 64 + n] : (_Float16)0.f; }
  else if (i < 14336) { int j = i - 10240; int n = j / 64,  k = j % 64;  w[i] = (_Float16)We2[k * 64 + n]; }
  else if (i < 18432) { int j = i - 14336; int n = j / 64,  k = j % 64;  w[i] = (_Float16)Wc1[k * 64 + n]; }
  else if (i < 26624) { int j = i - 18432; int n = j / 128, k = j % 128; w[i] = (_Float16)Wn1[k * 64 + n]; }
  else if (i < 30720) { int j = i - 26624; int n = j / 64,  k = j % 64;  w[i] = (_Float16)Wn2[k * 64 + n]; }
}

// ---------- edge kernel: edge MLP + coord gate + scatters ----------

__global__ __launch_bounds__(256) void segno_edge_kernel(
    const float* __restrict__ h, const float* __restrict__ coord,
    const int* __restrict__ eidx, int E,
    const _Float16* __restrict__ wt1, const _Float16* __restrict__ wt2,
    const _Float16* __restrict__ wc1t,
    const float* __restrict__ be1, const float* __restrict__ be2,
    const float* __restrict__ bc1, const float* __restrict__ Wc2,
    const float* __restrict__ bc2,
    float* __restrict__ aggh, float* __restrict__ sumt, float* __restrict__ cntp) {
  __shared__ _Float16 A1[EPB][A1_STRIDE];   // [h_src | h_dst | radial | 0-pad], K=160
  __shared__ _Float16 HB[EPB][HB_STRIDE];   // hidden / edge_feat / gate-hidden (reused)
  __shared__ int   rsrc[EPB];
  __shared__ float cdiff[EPB][3];

  int tid = threadIdx.x, lane = tid & 31, wave = tid >> 5;
  int ebase = blockIdx.x * EPB;

  { // stage one half-row (64 f16) per thread; wave-local rows
    int e = tid >> 1, hi = tid & 1;
    int ge = ebase + e;
    _Float16* ap = &A1[e][hi * F];
    if (ge < E) {
      int src = eidx[ge], dst = eidx[E + ge];
      const float* hp = h + (size_t)(hi ? dst : src) * F;
#pragma unroll
      for (int k = 0; k < F; k += 8) {
        float4 f0 = *(const float4*)(hp + k);
        float4 f1 = *(const float4*)(hp + k + 4);
        v8h o = { (_Float16)f0.x, (_Float16)f0.y, (_Float16)f0.z, (_Float16)f0.w,
                  (_Float16)f1.x, (_Float16)f1.y, (_Float16)f1.z, (_Float16)f1.w };
        *(v8h*)(ap + k) = o;
      }
      if (hi == 0) {
        rsrc[e] = src;
        float dx = coord[src * 3 + 0] - coord[dst * 3 + 0];
        float dy = coord[src * 3 + 1] - coord[dst * 3 + 1];
        float dz = coord[src * 3 + 2] - coord[dst * 3 + 2];
        cdiff[e][0] = dx; cdiff[e][1] = dy; cdiff[e][2] = dz;
        v8h z = {};
        *(v8h*)&A1[e][128] = z; *(v8h*)&A1[e][136] = z;
        *(v8h*)&A1[e][144] = z; *(v8h*)&A1[e][152] = z;
        A1[e][128] = (_Float16)(dx * dx + dy * dy + dz * dz);
      }
    } else {
      v8h z = {};
#pragma unroll
      for (int k = 0; k < F; k += 8) *(v8h*)(ap + k) = z;
      if (hi == 0) {
        rsrc[e] = -1;
        cdiff[e][0] = 0.f; cdiff[e][1] = 0.f; cdiff[e][2] = 0.f;
        *(v8h*)&A1[e][128] = z; *(v8h*)&A1[e][136] = z;
        *(v8h*)&A1[e][144] = z; *(v8h*)&A1[e][152] = z;
      }
    }
  }
  __syncthreads();

  const _Float16* Abase = &A1[wave * 16][0];
  _Float16* Hbase = &HB[wave * 16][0];
  int m0 = wave * 16;

  // edge_mlp layer 1: [16x160] @ [160x64]  (20 WMMA)
  v8f acc[4];
#pragma unroll
  for (int t = 0; t < 4; t++) acc[t] = 0;
  gemm_tile<5>(Abase, A1_STRIDE, wt1, 160, lane, acc);
  relu_store(acc, be1, Hbase, HB_STRIDE, lane);

  // edge_mlp layer 2 -> edge_feat  (8 WMMA)
#pragma unroll
  for (int t = 0; t < 4; t++) acc[t] = 0;
  gemm_tile<2>(Hbase, HB_STRIDE, wt2, 64, lane, acc);

  { // ReLU, scatter-add into agg_h, keep f16 copy in HB for the gate GEMM
    int g = lane >> 4, nl = lane & 15;
#pragma unroll
    for (int t = 0; t < 4; t++) {
      int n = nl + 16 * t;
      float bb = be2[n];
#pragma unroll
      for (int i = 0; i < 8; i++) {
        float v = acc[t][i] + bb;
        v = v > 0.f ? v : 0.f;
        int m = m0 + i + 8 * g;
        Hbase[(i + 8 * g) * HB_STRIDE + n] = (_Float16)v;
        int r = rsrc[m];
        if (r >= 0) atomicAdd(&aggh[(size_t)r * F + n], v);
      }
    }
  }

  // coord gate hidden: relu(ef @ Wc1 + bc1)  (8 WMMA)
#pragma unroll
  for (int t = 0; t < 4; t++) acc[t] = 0;
  gemm_tile<2>(Hbase, HB_STRIDE, wc1t, 64, lane, acc);
  relu_store(acc, bc1, Hbase, HB_STRIDE, lane);

  // gate = hidden . Wc2 + bc2 ; trans = clip(coord_diff*gate); scatter
  if (lane < 16) {
    int m = m0 + lane;
    int r = rsrc[m];
    if (r >= 0) {
      float gsum = bc2[0];
#pragma unroll 8
      for (int k = 0; k < 64; k++) gsum += (float)HB[m][k] * Wc2[k];
      float tx = cdiff[m][0] * gsum, ty = cdiff[m][1] * gsum, tz = cdiff[m][2] * gsum;
      tx = fminf(fmaxf(tx, -100.f), 100.f);
      ty = fminf(fmaxf(ty, -100.f), 100.f);
      tz = fminf(fmaxf(tz, -100.f), 100.f);
      atomicAdd(&sumt[(size_t)r * 3 + 0], tx);
      atomicAdd(&sumt[(size_t)r * 3 + 1], ty);
      atomicAdd(&sumt[(size_t)r * 3 + 2], tz);
      atomicAdd(&cntp[r], 1.0f);
    }
  }
}

// ---------- node kernel: node MLP + residual + coord/vel finalize ----------

__global__ __launch_bounds__(256) void segno_node_kernel(
    const float* __restrict__ h, const float* __restrict__ aggh,
    const float* __restrict__ sumt, const float* __restrict__ cntp,
    const float* __restrict__ coord, const float* __restrict__ vel,
    const _Float16* __restrict__ wn1t, const _Float16* __restrict__ wn2t,
    const float* __restrict__ bn1, const float* __restrict__ bn2, int N,
    float* __restrict__ hout, float* __restrict__ coordOut, float* __restrict__ vOut) {
  __shared__ _Float16 A[EPB][AN_STRIDE];    // [h | agg_h], K=128
  __shared__ _Float16 HB[EPB][HB_STRIDE];
  int tid = threadIdx.x, lane = tid & 31, wave = tid >> 5;
  int nbase = blockIdx.x * EPB;

  { // stage
    int r = tid >> 1, hi = tid & 1;
    int node = nbase + r;
    _Float16* ap = &A[r][hi * F];
    if (node < N) {
      const float* sp = (hi ? aggh : h) + (size_t)node * F;
#pragma unroll
      for (int k = 0; k < F; k += 8) {
        float4 f0 = *(const float4*)(sp + k);
        float4 f1 = *(const float4*)(sp + k + 4);
        v8h o = { (_Float16)f0.x, (_Float16)f0.y, (_Float16)f0.z, (_Float16)f0.w,
                  (_Float16)f1.x, (_Float16)f1.y, (_Float16)f1.z, (_Float16)f1.w };
        *(v8h*)(ap + k) = o;
      }
    } else {
      v8h z = {};
#pragma unroll
      for (int k = 0; k < F; k += 8) *(v8h*)(ap + k) = z;
    }
  }

  // coord / velocity finalize (segment mean, integrate)
  if (tid < EPB) {
    int node = nbase + tid;
    if (node < N) {
      const float invL = 0.125f;
      float c = cntp[node];
      float den = fmaxf(c, 1.0f);
      float scale = (c > 0.f) ? (invL / den) : 0.f;   // COORDS_WEIGHT = 1
      float vx = vel[node * 3 + 0] + sumt[node * 3 + 0] * scale;
      float vy = vel[node * 3 + 1] + sumt[node * 3 + 1] * scale;
      float vz = vel[node * 3 + 2] + sumt[node * 3 + 2] * scale;
      vOut[node * 3 + 0] = vx; vOut[node * 3 + 1] = vy; vOut[node * 3 + 2] = vz;
      coordOut[node * 3 + 0] = coord[node * 3 + 0] + vx * invL;
      coordOut[node * 3 + 1] = coord[node * 3 + 1] + vy * invL;
      coordOut[node * 3 + 2] = coord[node * 3 + 2] + vz * invL;
    }
  }
  __syncthreads();

  const _Float16* Abase = &A[wave * 16][0];
  _Float16* Hbase = &HB[wave * 16][0];

  v8f acc[4];
#pragma unroll
  for (int t = 0; t < 4; t++) acc[t] = 0;
  gemm_tile<4>(Abase, AN_STRIDE, wn1t, 128, lane, acc);   // 16 WMMA
  relu_store(acc, bn1, Hbase, HB_STRIDE, lane);

#pragma unroll
  for (int t = 0; t < 4; t++) acc[t] = 0;
  gemm_tile<2>(Hbase, HB_STRIDE, wn2t, 64, lane, acc);    // 8 WMMA

  { // residual + bias, store h_out
    int g = lane >> 4, nl = lane & 15, m0 = wave * 16;
#pragma unroll
    for (int t = 0; t < 4; t++) {
      int n = nl + 16 * t;
      float bb = bn2[n];
#pragma unroll
      for (int i = 0; i < 8; i++) {
        int node = nbase + m0 + i + 8 * g;
        if (node < N)
          hout[(size_t)node * F + n] = h[(size_t)node * F + n] + acc[t][i] + bb;
      }
    }
  }
}

// ---------- launcher ----------

extern "C" void kernel_launch(void* const* d_in, const int* in_sizes, int n_in,
                              void* d_out, int out_size, void* d_ws, size_t ws_size,
                              hipStream_t stream) {
  const float* h     = (const float*)d_in[0];
  const float* coord = (const float*)d_in[1];
  const float* vel   = (const float*)d_in[2];
  // d_in[3] = vel_init (unused by reference forward)
  const int*   eidx  = (const int*)d_in[4];
  const float* We1 = (const float*)d_in[5];  const float* be1 = (const float*)d_in[6];
  const float* We2 = (const float*)d_in[7];  const float* be2 = (const float*)d_in[8];
  const float* Wn1 = (const float*)d_in[9];  const float* bn1 = (const float*)d_in[10];
  const float* Wn2 = (const float*)d_in[11]; const float* bn2 = (const float*)d_in[12];
  const float* Wc1 = (const float*)d_in[13]; const float* bc1 = (const float*)d_in[14];
  const float* Wc2 = (const float*)d_in[15]; const float* bc2 = (const float*)d_in[16];

  int N = in_sizes[0] / F;
  int E = in_sizes[4] / 2;

  char* ws = (char*)d_ws;
  float* aggh = (float*)ws;                 // [N][64]
  float* sumt = aggh + (size_t)N * F;       // [N][3]
  float* cntp = sumt + (size_t)N * 3;       // [N]
  size_t zcount  = (size_t)N * (F + 4);     // all f32 accumulators
  size_t halfOff = ((size_t)N * (F + 4) * 4 + 31) & ~(size_t)31;
  _Float16* wh = (_Float16*)(ws + halfOff); // f16 weight arena (30720 halfs)

  segno_zero_ws<<<2048, 256, 0, stream>>>(aggh, (int)zcount);
  segno_prep_w<<<(30720 + 255) / 256, 256, 0, stream>>>(We1, We2, Wc1, Wn1, Wn2, wh);

  const _Float16* wt1  = wh;
  const _Float16* wt2  = wh + 10240;
  const _Float16* wc1t = wh + 14336;
  const _Float16* wn1t = wh + 18432;
  const _Float16* wn2t = wh + 26624;

  float* hout     = (float*)d_out;
  float* coordOut = hout + (size_t)N * F;
  float* vOut     = coordOut + (size_t)N * 3;

  segno_edge_kernel<<<(E + EPB - 1) / EPB, 256, 0, stream>>>(
      h, coord, eidx, E, wt1, wt2, wc1t, be1, be2, bc1, Wc2, bc2, aggh, sumt, cntp);
  segno_node_kernel<<<(N + EPB - 1) / EPB, 256, 0, stream>>>(
      h, aggh, sumt, cntp, coord, vel, wn1t, wn2t, bn1, bn2, N, hout, coordOut, vOut);
}
